// rnn_gat_44495861187265
// MI455X (gfx1250) — matrix operand order repaired and checked by reading, not compiled, not run on hardware
//
#include <hip/hip_runtime.h>
#include <hip/hip_bf16.h>

// ---------------------------------------------------------------------------
// RNN-GAT fused pipeline for MI455X (gfx1250, wave32, WMMA 16x16x32 bf16)
// ---------------------------------------------------------------------------

typedef __attribute__((ext_vector_type(16))) __bf16 v16bf;
typedef __attribute__((ext_vector_type(8)))  __bf16 v8bf;
typedef __attribute__((ext_vector_type(8)))  float  v8f;

#define V8F_ZERO {0.f,0.f,0.f,0.f,0.f,0.f,0.f,0.f}

__device__ __forceinline__ v8f wmma_bf16(v16bf a, v16bf b, v8f c) {
  return __builtin_amdgcn_wmma_f32_16x16x32_bf16(false, a, false, b, (short)0, c,
                                                 false, false);
}

// A fragment (16x32 bf16, MxK). base points at row0/col0; rowstride in halfs.
// Lane L holds row m=L&15, K in [koff,koff+16), koff=(L>=16)*16.
__device__ __forceinline__ v16bf load_a_frag(const __bf16* base, int rowstride,
                                             int kbase, int lane) {
  int m    = lane & 15;
  int koff = (lane >> 4) << 4;
  const __bf16* p = base + (size_t)m * rowstride + kbase + koff;
  v8bf lo = *(const v8bf*)(p);
  v8bf hi = *(const v8bf*)(p + 8);
  v16bf f;
#pragma unroll
  for (int j = 0; j < 8; ++j) { f[j] = lo[j]; f[8 + j] = hi[j]; }
  return f;
}

// B fragment (32x16 bf16, KxN) from weights stored [outN][Kpad] (row-major).
// Lane L holds column n=L&15, K = kbase + (L>=16)*16 + j  -> 32 contiguous bytes.
__device__ __forceinline__ v16bf load_b_frag(const __bf16* W, int kpad,
                                             int ntile, int kbase, int lane) {
  int n    = lane & 15;
  int koff = (lane >> 4) << 4;
  const __bf16* p = W + (size_t)(ntile * 16 + n) * kpad + kbase + koff;
  return *(const v16bf*)p;
}

__device__ __forceinline__ float sigmoidf_(float x) {
  return 1.f / (1.f + __expf(-x));
}
__device__ __forceinline__ float tanhf_(float x) {
  float e = __expf(-2.f * fabsf(x));
  float t = (1.f - e) / (1.f + e);
  return x >= 0.f ? t : -t;
}
// Order-preserving float<->uint encoding for atomicMax on floats.
__device__ __forceinline__ unsigned fenc(float f) {
  unsigned u = __float_as_uint(f);
  return (u & 0x80000000u) ? ~u : (u | 0x80000000u);
}
__device__ __forceinline__ float fdec(unsigned u) {
  return (u & 0x80000000u) ? __uint_as_float(u & 0x7FFFFFFFu)
                           : __uint_as_float(~u);
}

// ---------------------------------------------------------------------------
__global__ void zero_kernel(float* p, size_t n) {
  size_t i = (size_t)blockIdx.x * blockDim.x + threadIdx.x;
  if (i < n) p[i] = 0.f;
}

// Reformat weights to bf16 padded-K layouts + bias sums.
__global__ void prep_weights_kernel(const float* Wih_n, const float* Whh_n,
                                    const float* bih_n, const float* bhh_n,
                                    const float* Wih_e, const float* Whh_e,
                                    const float* bih_e, const float* bhh_e,
                                    const float* W_gat,
                                    __bf16* Wce, __bf16* Wcn, __bf16* Wgt,
                                    float* b_e, float* b_n) {
  int i = blockIdx.x * blockDim.x + threadIdx.x;
  if (i < 128 * 64) {  // edge: [128][2+32 -> 64]
    int r = i >> 6, c = i & 63;
    float v = 0.f;
    if (c < 2)       v = Wih_e[r * 2 + c];
    else if (c < 34) v = Whh_e[r * 32 + (c - 2)];
    Wce[i] = (__bf16)v;
  }
  if (i < 512 * 160) {  // node: [512][5+128 -> 160]
    int r = i / 160, c = i % 160;
    float v = 0.f;
    if (c < 5)        v = Wih_n[r * 5 + c];
    else if (c < 133) v = Whh_n[r * 128 + (c - 5)];
    Wcn[i] = (__bf16)v;
  }
  if (i < 64 * 160) {  // GAT: transpose [160][64] -> [64][160]
    int o = i / 160, k = i % 160;
    Wgt[i] = (__bf16)W_gat[(size_t)k * 64 + o];
  }
  if (i < 128) b_e[i] = bih_e[i] + bhh_e[i];
  if (i < 512) b_n[i] = bih_n[i] + bhh_n[i];
}

// ---------------------------------------------------------------------------
// Edge LSTM: gates[E,128] = [edge_attr | h_edge_h](pad K=64) x Wce^T, fused
// activations, he/ce stores, scatter-add of he into esum[dst] + degree.
#define WAVES_E 8
__global__ void edge_lstm_kernel(const float* __restrict__ edge_attr,
                                 const int* __restrict__ edge_index,
                                 const float* __restrict__ h_edge_h,
                                 const float* __restrict__ h_edge_c,
                                 const __bf16* __restrict__ Wce,
                                 const float* __restrict__ b_e,
                                 float* __restrict__ he_out,
                                 float* __restrict__ ce_out,
                                 float* __restrict__ esum,
                                 float* __restrict__ deg, int E) {
  __shared__ __align__(16) __bf16 sA[WAVES_E][16 * 64];
  int wave = threadIdx.x >> 5;
  int lane = threadIdx.x & 31;
  int tile = blockIdx.x * WAVES_E + wave;
  bool valid = (tile * 16) < E;
  int e0 = tile * 16;

  {  // stage A tile: lane covers row r=lane>>1, 32 cols
    int r = lane >> 1;
    int c0 = (lane & 1) * 32;
    __bf16* dp = &sA[wave][r * 64 + c0];
    if (valid) {
      int e = e0 + r;
#pragma unroll
      for (int c = 0; c < 32; ++c) {
        int cc = c0 + c;
        float v = 0.f;
        if (cc < 2)       v = edge_attr[(size_t)e * 2 + cc];
        else if (cc < 34) v = h_edge_h[(size_t)e * 32 + (cc - 2)];
        dp[c] = (__bf16)v;
      }
    } else {
      for (int c = 0; c < 32; ++c) dp[c] = (__bf16)0.f;
    }
  }
  __syncthreads();

  __builtin_prefetch(Wce, 0, 3);
  v16bf a0 = load_a_frag(&sA[wave][0], 64, 0, lane);
  v16bf a1 = load_a_frag(&sA[wave][0], 64, 32, lane);

  v8f acc[8];
#pragma unroll
  for (int nt = 0; nt < 8; ++nt) {
    v8f c = V8F_ZERO;
    c = wmma_bf16(a0, load_b_frag(Wce, 64, nt, 0, lane), c);
    c = wmma_bf16(a1, load_b_frag(Wce, 64, nt, 32, lane), c);
    acc[nt] = c;
  }

  if (valid) {
    int n = lane & 15;
    int hi = lane >> 4;
#pragma unroll
    for (int p = 0; p < 2; ++p) {
      float bi = b_e[p * 16 + n];
      float bf = b_e[32 + p * 16 + n];
      float bg = b_e[64 + p * 16 + n];
      float bo = b_e[96 + p * 16 + n];
#pragma unroll
      for (int v = 0; v < 8; ++v) {
        int m = hi * 8 + v;
        int e = e0 + m;
        float gi = sigmoidf_(acc[p][v] + bi);
        float gf = sigmoidf_(acc[p + 2][v] + bf);
        float gg = tanhf_(acc[p + 4][v] + bg);
        float go = sigmoidf_(acc[p + 6][v] + bo);
        float cold = h_edge_c[(size_t)e * 32 + p * 16 + n];
        float c2 = gf * cold + gi * gg;
        float hv = go * tanhf_(c2);
        size_t off = (size_t)e * 32 + p * 16 + n;
        he_out[off] = hv;
        ce_out[off] = c2;
        int d = edge_index[(size_t)E + e];
        atomicAdd(&esum[(size_t)d * 32 + p * 16 + n], hv);
        if (p == 0 && n == 0) atomicAdd(&deg[d], 1.0f);
      }
    }
  }
}

// ---------------------------------------------------------------------------
// Node LSTM: gates[N,512] = [x | h_node_h](pad K=160) x Wcn^T, fused
// activations; writes hn/cn and bf16 feat[:,0:128].
#define WAVES_N 4
__global__ void node_lstm_kernel(const float* __restrict__ x,
                                 const float* __restrict__ h_node_h,
                                 const float* __restrict__ h_node_c,
                                 const __bf16* __restrict__ Wcn,
                                 const float* __restrict__ b_n,
                                 float* __restrict__ hn_out,
                                 float* __restrict__ cn_out,
                                 __bf16* __restrict__ featbf, int N) {
  __shared__ __align__(16) __bf16 sA[WAVES_N][16 * 160];
  int wave = threadIdx.x >> 5;
  int lane = threadIdx.x & 31;
  int tile = blockIdx.x * WAVES_N + wave;
  bool valid = (tile * 16) < N;
  int n0 = tile * 16;

  {  // stage A tile: lane covers row r=lane>>1, 80 cols
    int r = lane >> 1;
    int c0 = (lane & 1) * 80;
    __bf16* dp = &sA[wave][r * 160 + c0];
    if (valid) {
      int node = n0 + r;
      for (int c = 0; c < 80; ++c) {
        int cc = c0 + c;
        float v = 0.f;
        if (cc < 5)        v = x[(size_t)node * 5 + cc];
        else if (cc < 133) v = h_node_h[(size_t)node * 128 + (cc - 5)];
        dp[c] = (__bf16)v;
      }
    } else {
      for (int c = 0; c < 80; ++c) dp[c] = (__bf16)0.f;
    }
  }
  __syncthreads();

  __builtin_prefetch(Wcn, 0, 3);
  v16bf a[5];
#pragma unroll
  for (int ks = 0; ks < 5; ++ks)
    a[ks] = load_a_frag(&sA[wave][0], 160, ks * 32, lane);

  int nn = lane & 15;
  int hi = lane >> 4;
  for (int jt = 0; jt < 8; ++jt) {  // feature tiles of 16 (of 128)
    v8f ai = V8F_ZERO, af = V8F_ZERO, ag = V8F_ZERO, ao = V8F_ZERO;
#pragma unroll
    for (int ks = 0; ks < 5; ++ks) {
      ai = wmma_bf16(a[ks], load_b_frag(Wcn, 160, jt,      ks * 32, lane), ai);
      af = wmma_bf16(a[ks], load_b_frag(Wcn, 160, jt + 8,  ks * 32, lane), af);
      ag = wmma_bf16(a[ks], load_b_frag(Wcn, 160, jt + 16, ks * 32, lane), ag);
      ao = wmma_bf16(a[ks], load_b_frag(Wcn, 160, jt + 24, ks * 32, lane), ao);
    }
    if (valid) {
      int j = jt * 16 + nn;
      float bi = b_n[j];
      float bfv = b_n[128 + j];
      float bg = b_n[256 + j];
      float bo = b_n[384 + j];
#pragma unroll
      for (int v = 0; v < 8; ++v) {
        int node = n0 + hi * 8 + v;
        float gi = sigmoidf_(ai[v] + bi);
        float gf = sigmoidf_(af[v] + bfv);
        float gg = tanhf_(ag[v] + bg);
        float go = sigmoidf_(ao[v] + bo);
        float cold = h_node_c[(size_t)node * 128 + j];
        float c2 = gf * cold + gi * gg;
        float hv = go * tanhf_(c2);
        hn_out[(size_t)node * 128 + j] = hv;
        cn_out[(size_t)node * 128 + j] = c2;
        featbf[(size_t)node * 160 + j] = (__bf16)hv;
      }
    }
  }
}

// edge_enc = esum/max(deg,1) -> bf16 feat[:,128:160]
__global__ void edge_enc_feat_kernel(const float* __restrict__ esum,
                                     const float* __restrict__ deg,
                                     __bf16* __restrict__ featbf, int N) {
  int i = blockIdx.x * blockDim.x + threadIdx.x;
  if (i < N * 32) {
    int node = i >> 5, k = i & 31;
    float d = deg[node];
    d = d > 1.f ? d : 1.f;
    featbf[(size_t)node * 160 + 128 + k] = (__bf16)(esum[i] / d);
  }
}

// ---------------------------------------------------------------------------
// GAT linear: xl[N,64] = feat[N,160] x WgatT^T (A direct from global bf16).
#define WAVES_G 8
__global__ void gat_gemm_kernel(const __bf16* __restrict__ featbf,
                                const __bf16* __restrict__ Wgt,
                                float* __restrict__ xl, int N) {
  int wave = threadIdx.x >> 5;
  int lane = threadIdx.x & 31;
  int tile = blockIdx.x * WAVES_G + wave;
  if (tile * 16 >= N) return;  // wave-uniform; EXEC stays all-ones for WMMA
  int n0 = tile * 16;

  __builtin_prefetch(Wgt, 0, 3);
  v16bf a[5];
#pragma unroll
  for (int ks = 0; ks < 5; ++ks)
    a[ks] = load_a_frag(featbf + (size_t)n0 * 160, 160, ks * 32, lane);

  int nn = lane & 15;
  int hi = lane >> 4;
#pragma unroll
  for (int h = 0; h < 4; ++h) {
    v8f acc = V8F_ZERO;
#pragma unroll
    for (int ks = 0; ks < 5; ++ks)
      acc = wmma_bf16(a[ks], load_b_frag(Wgt, 160, h, ks * 32, lane), acc);
#pragma unroll
    for (int v = 0; v < 8; ++v)
      xl[(size_t)(n0 + hi * 8 + v) * 64 + h * 16 + nn] = acc[v];
  }
}

// per-node attention coefficients a_src/a_dst [N,4]
__global__ void attn_coef_kernel(const float* __restrict__ xl,
                                 const float* __restrict__ att_src,
                                 const float* __restrict__ att_dst,
                                 float* __restrict__ a_src,
                                 float* __restrict__ a_dst, int N) {
  int i = blockIdx.x * blockDim.x + threadIdx.x;
  if (i < N * 4) {
    int node = i >> 2, h = i & 3;
    const float* xp = xl + (size_t)node * 64 + h * 16;
    float s = 0.f, d = 0.f;
#pragma unroll
    for (int c = 0; c < 16; ++c) {
      s += xp[c] * att_src[h * 16 + c];
      d += xp[c] * att_dst[h * 16 + c];
    }
    a_src[i] = s;
    a_dst[i] = d;
  }
}

// pass1: alpha = leaky_relu(a_src[src]+a_dst[dst]); segment-max via atomicMax.
__global__ void attn_max_kernel(const int* __restrict__ ei,
                                const float* __restrict__ a_src,
                                const float* __restrict__ a_dst,
                                float* __restrict__ alpha,
                                unsigned* __restrict__ mmax, int E, int N) {
  int ae = blockIdx.x * blockDim.x + threadIdx.x;
  if (ae < E + N) {
    int s, d;
    if (ae < E) { s = ei[ae]; d = ei[(size_t)E + ae]; }
    else        { s = d = ae - E; }
#pragma unroll
    for (int h = 0; h < 4; ++h) {
      float a = a_src[(size_t)s * 4 + h] + a_dst[(size_t)d * 4 + h];
      a = a > 0.f ? a : 0.2f * a;
      alpha[(size_t)ae * 4 + h] = a;
      atomicMax(&mmax[(size_t)d * 4 + h], fenc(a));
    }
  }
}

// pass2: ea = exp(alpha - m[dst]); segment-sum via atomicAdd.
__global__ void attn_exp_kernel(const int* __restrict__ ei,
                                const unsigned* __restrict__ mmax,
                                float* __restrict__ alpha,
                                float* __restrict__ ssum, int E, int N) {
  int ae = blockIdx.x * blockDim.x + threadIdx.x;
  if (ae < E + N) {
    int d = (ae < E) ? ei[(size_t)E + ae] : (ae - E);
#pragma unroll
    for (int h = 0; h < 4; ++h) {
      float e = __expf(alpha[(size_t)ae * 4 + h] - fdec(mmax[(size_t)d * 4 + h]));
      alpha[(size_t)ae * 4 + h] = e;
      atomicAdd(&ssum[(size_t)d * 4 + h], e);
    }
  }
}

// pass3: outacc[dst,h,:] += xl[src,h,:] * ea   (thread per (edge,head))
__global__ void attn_agg_kernel(const int* __restrict__ ei,
                                const float* __restrict__ alpha,
                                const float* __restrict__ xl,
                                float* __restrict__ outacc, int E, int N) {
  int t = blockIdx.x * blockDim.x + threadIdx.x;
  if (t < (E + N) * 4) {
    int ae = t >> 2, h = t & 3;
    int s, d;
    if (ae < E) { s = ei[ae]; d = ei[(size_t)E + ae]; }
    else        { s = d = ae - E; }
    float w = alpha[(size_t)ae * 4 + h];
    const float* xp = xl + (size_t)s * 64 + h * 16;
    float* op = outacc + (size_t)d * 64 + h * 16;
#pragma unroll
    for (int c = 0; c < 16; ++c) atomicAdd(&op[c], xp[c] * w);
  }
}

// finalize: out = mean_h(outacc/ssum) + bias
__global__ void out_final_kernel(const float* __restrict__ outacc,
                                 const float* __restrict__ ssum,
                                 const float* __restrict__ bias,
                                 float* __restrict__ out, int N) {
  int i = blockIdx.x * blockDim.x + threadIdx.x;
  if (i < N * 16) {
    int node = i >> 4, c = i & 15;
    float acc = 0.f;
#pragma unroll
    for (int h = 0; h < 4; ++h)
      acc += outacc[(size_t)node * 64 + h * 16 + c] / ssum[(size_t)node * 4 + h];
    out[i] = 0.25f * acc + bias[c];
  }
}

// ---------------------------------------------------------------------------
extern "C" void kernel_launch(void* const* d_in, const int* in_sizes, int n_in,
                              void* d_out, int out_size, void* d_ws,
                              size_t ws_size, hipStream_t stream) {
  const float* x        = (const float*)d_in[0];
  const int*   ei       = (const int*)d_in[1];
  const float* eattr    = (const float*)d_in[2];
  const float* hnh      = (const float*)d_in[3];
  const float* hnc      = (const float*)d_in[4];
  const float* heh      = (const float*)d_in[5];
  const float* hec      = (const float*)d_in[6];
  const float* Wih_n    = (const float*)d_in[7];
  const float* Whh_n    = (const float*)d_in[8];
  const float* bih_n    = (const float*)d_in[9];
  const float* bhh_n    = (const float*)d_in[10];
  const float* Wih_e    = (const float*)d_in[11];
  const float* Whh_e    = (const float*)d_in[12];
  const float* bih_e    = (const float*)d_in[13];
  const float* bhh_e    = (const float*)d_in[14];
  const float* W_gat    = (const float*)d_in[15];
  const float* att_src  = (const float*)d_in[16];
  const float* att_dst  = (const float*)d_in[17];
  const float* bias_gat = (const float*)d_in[18];

  const int N = in_sizes[0] / 5;   // node count
  const int E = in_sizes[2] / 2;   // edge count

  // d_out layout: out[N,16] | hn[N,128] | cn[N,128] | he[E,32] | ce[E,32]
  float* out = (float*)d_out;
  float* hn  = out + (size_t)N * 16;
  float* cn  = hn + (size_t)N * 128;
  float* he  = cn + (size_t)N * 128;
  float* ce  = he + (size_t)E * 32;

  // workspace layout (f32 elements)
  float* ws = (float*)d_ws;
  size_t off = 0;
  float*    esum   = ws + off; off += (size_t)N * 32;
  float*    deg    = ws + off; off += (size_t)N;
  float*    ssum   = ws + off; off += (size_t)N * 4;
  float*    outacc = ws + off; off += (size_t)N * 64;
  unsigned* mmax   = (unsigned*)(ws + off); off += (size_t)N * 4;
  size_t zero_cnt  = off;                       // esum..mmax zeroed together
  float*    xl     = ws + off; off += (size_t)N * 64;
  float*    a_src  = ws + off; off += (size_t)N * 4;
  float*    a_dst  = ws + off; off += (size_t)N * 4;
  float*    alpha  = ws + off; off += (size_t)(E + N) * 4;
  float*    b_e    = ws + off; off += 128;
  float*    b_n    = ws + off; off += 512;
  size_t bfbase = (off * sizeof(float) + 127) & ~(size_t)127;
  __bf16* bf = (__bf16*)((char*)d_ws + bfbase);
  __bf16* Wce    = bf;                         // 128*64
  __bf16* Wcn    = Wce + 128 * 64;             // 512*160
  __bf16* Wgt    = Wcn + 512 * 160;            // 64*160
  __bf16* featbf = Wgt + 64 * 160;             // N*160

  const int TB = 256;

  // 1) zero accumulators
  zero_kernel<<<(int)((zero_cnt + TB - 1) / TB), TB, 0, stream>>>(ws, zero_cnt);

  // 2) weight reformat (bf16 padded-K) + bias sums
  prep_weights_kernel<<<(512 * 160 + TB - 1) / TB, TB, 0, stream>>>(
      Wih_n, Whh_n, bih_n, bhh_n, Wih_e, Whh_e, bih_e, bhh_e, W_gat,
      Wce, Wcn, Wgt, b_e, b_n);

  // 3) edge LSTM (WMMA) + scatter-add of he into esum/deg
  {
    int tiles = (E + 15) / 16;
    int blocks = (tiles + WAVES_E - 1) / WAVES_E;
    edge_lstm_kernel<<<blocks, WAVES_E * 32, 0, stream>>>(
        eattr, ei, heh, hec, Wce, b_e, he, ce, esum, deg, E);
  }

  // 4) node LSTM (WMMA) -> hn, cn, feat[:,0:128]
  {
    int tiles = (N + 15) / 16;
    int blocks = (tiles + WAVES_N - 1) / WAVES_N;
    node_lstm_kernel<<<blocks, WAVES_N * 32, 0, stream>>>(
        x, hnh, hnc, Wcn, b_n, hn, cn, featbf, N);
  }

  // 5) edge_enc mean -> feat[:,128:160]
  edge_enc_feat_kernel<<<(N * 32 + TB - 1) / TB, TB, 0, stream>>>(esum, deg,
                                                                  featbf, N);

  // 6) GAT linear (WMMA) -> xl[N,64]
  {
    int tiles = (N + 15) / 16;
    int blocks = (tiles + WAVES_G - 1) / WAVES_G;
    gat_gemm_kernel<<<blocks, WAVES_G * 32, 0, stream>>>(featbf, Wgt, xl, N);
  }

  // 7) attention coefficients
  attn_coef_kernel<<<(N * 4 + TB - 1) / TB, TB, 0, stream>>>(xl, att_src,
                                                             att_dst, a_src,
                                                             a_dst, N);

  // 8-10) segment softmax + weighted aggregation over edges (+self loops)
  int nae = E + N;
  attn_max_kernel<<<(nae + TB - 1) / TB, TB, 0, stream>>>(ei, a_src, a_dst,
                                                          alpha, mmax, E, N);
  attn_exp_kernel<<<(nae + TB - 1) / TB, TB, 0, stream>>>(ei, mmax, alpha,
                                                          ssum, E, N);
  attn_agg_kernel<<<(nae * 4 + TB - 1) / TB, TB, 0, stream>>>(ei, alpha, xl,
                                                              outacc, E, N);

  // 11) head-mean + bias
  out_final_kernel<<<(N * 16 + TB - 1) / TB, TB, 0, stream>>>(outacc, ssum,
                                                              bias_gat, out, N);
}
